// MovementPrunedLinear_48344151883970
// MI455X (gfx1250) — matrix operand order repaired and checked by reading, not compiled, not run on hardware
//
#include <hip/hip_runtime.h>

// ---- WMMA fragment types (gfx1250 wave32) ----------------------------------
typedef __attribute__((ext_vector_type(16))) __bf16        v16bf;
typedef __attribute__((ext_vector_type(8)))  float         v8f;
typedef __attribute__((ext_vector_type(4)))  unsigned int  v4u;

union Frag32B { v16bf bf; v4u q[2]; };   // 32 bytes: 16 bf16 = 8 VGPRs

#define LOGIT_TAU (-2.1972246f)          // log(0.1/0.9): sigmoid(s)>0.1 <=> s>LOGIT_TAU

__device__ __forceinline__ unsigned pack_bf16x2(float a, float b) {
    __bf16 lo = (__bf16)a;
    __bf16 hi = (__bf16)b;
    unsigned short ulo = __builtin_bit_cast(unsigned short, lo);
    unsigned short uhi = __builtin_bit_cast(unsigned short, hi);
    return (unsigned)ulo | ((unsigned)uhi << 16);
}

// CDNA5 async copy: global -> LDS, 16 bytes per lane, tracked by ASYNCcnt.
__device__ __forceinline__ void async_ld_b128(unsigned lds_addr, const void* gaddr) {
    asm volatile("global_load_async_to_lds_b128 %0, %1, off"
                 :: "v"(lds_addr), "v"(gaddr) : "memory");
}

// ---- Kernel 1: fused block-mask + f32->bf16 convert of W -------------------
__global__ __launch_bounds__(256) void prune_convert_w(
    const float* __restrict__ w, const float* __restrict__ scores,
    unsigned short* __restrict__ wb, int K, int KB /* = K/32 */) {
    int e = (blockIdx.x * 256 + threadIdx.x) * 8;
    int n = e / K;
    int k = e - n * K;
    v4u o = {0u, 0u, 0u, 0u};
    if (scores[(n >> 5) * KB + (k >> 5)] > LOGIT_TAU) {
        float4 f0 = *(const float4*)(w + e);
        float4 f1 = *(const float4*)(w + e + 4);
        o.x = pack_bf16x2(f0.x, f0.y);
        o.y = pack_bf16x2(f0.z, f0.w);
        o.z = pack_bf16x2(f1.x, f1.y);
        o.w = pack_bf16x2(f1.z, f1.w);
    }
    *(v4u*)(wb + e) = o;
}

// ---- Kernel 2: f32->bf16 convert of x --------------------------------------
__global__ __launch_bounds__(256) void convert_x(
    const float* __restrict__ x, unsigned short* __restrict__ xb) {
    int e = (blockIdx.x * 256 + threadIdx.x) * 8;
    float4 f0 = *(const float4*)(x + e);
    float4 f1 = *(const float4*)(x + e + 4);
    v4u o;
    o.x = pack_bf16x2(f0.x, f0.y);
    o.y = pack_bf16x2(f0.z, f0.w);
    o.z = pack_bf16x2(f1.x, f1.y);
    o.w = pack_bf16x2(f1.z, f1.w);
    *(v4u*)(xb + e) = o;
}

// ---- Kernel 3: bf16 WMMA GEMM: out[M,N] = xb[M,K] * wb[N,K]^T + bias -------
// 256 threads = 8 waves. Workgroup tile 256(M) x 128(N). Waves arranged
// 4(M) x 2(N); each wave computes 64x64 = 4x4 tiles of v_wmma_f32_16x16x32_bf16.
// K staged 64-wide through double-buffered LDS via global_load_async_to_lds.
#define ABUF_ELEMS (256 * 64)                 // 32 KB per buffer
#define BBUF_ELEMS (128 * 64)                 // 16 KB per buffer
#define BUF_STRIDE (ABUF_ELEMS + BBUF_ELEMS)  // 48 KB per buffer

__global__ __launch_bounds__(256) void wmma_gemm(
    const unsigned short* __restrict__ xb,   // [M,K] bf16
    const unsigned short* __restrict__ wb,   // [N,K] bf16 (pruned)
    const float* __restrict__ bias,          // [N]
    float* __restrict__ out,                 // [M,N]
    int M, int N, int K) {
    __shared__ unsigned short smem[2 * BUF_STRIDE];   // 96 KB double buffer

    const int m0   = blockIdx.y * 256;
    const int n0   = blockIdx.x * 128;
    const int tid  = threadIdx.x;
    const int wave = tid >> 5;
    const int lane = tid & 31;
    const int wm   = wave >> 1;   // 0..3 : which 64-row band
    const int wn   = wave & 1;    // 0..1 : which 64-col band
    const int h    = lane >> 4;   // lane half (VGPR-layout selector)
    const int r    = lane & 15;   // row / col within a 16x16 tile

    // ---- per-thread staging geometry (fixed across K iterations) -----------
    const int srow = tid >> 3;            // 0..31, + 32*i per chunk
    const int skc  = (tid & 7) << 3;      // 8-element (16 B) chunk within the 64-wide K slab
    const unsigned short* ga = xb + (size_t)(m0 + srow) * K + skc;
    const unsigned short* gb = wb + (size_t)(n0 + srow) * K + skc;
    const unsigned smem_base = (unsigned)(unsigned long long)(&smem[0]);
    const unsigned la0 = smem_base + (unsigned)(srow * 64 + skc) * 2;              // A chunk 0
    const unsigned lb0 = smem_base + (unsigned)(ABUF_ELEMS + srow * 64 + skc) * 2; // B chunk 0
    const size_t  rowstep = (size_t)32 * K;   // 32 rows per chunk step

    v8f acc[4][4];
    v8f zero = {0.f, 0.f, 0.f, 0.f, 0.f, 0.f, 0.f, 0.f};
#pragma unroll
    for (int i = 0; i < 4; ++i)
#pragma unroll
        for (int j = 0; j < 4; ++j) acc[i][j] = zero;

    // issue async stage of K-chunk `kb` into buffer `buf` (12 b128 per lane)
    auto stage = [&](int buf, int kb) {
        const unsigned short* pa = ga + kb;
        const unsigned short* pb = gb + kb;
        const unsigned lbuf = (unsigned)(buf * BUF_STRIDE * 2);
#pragma unroll
        for (int i = 0; i < 8; ++i)       // A: 256 rows
            async_ld_b128(la0 + lbuf + (unsigned)i * 4096, pa + (size_t)i * rowstep);
#pragma unroll
        for (int i = 0; i < 4; ++i)       // B: 128 rows
            async_ld_b128(lb0 + lbuf + (unsigned)i * 4096, pb + (size_t)i * rowstep);
    };

    const int nkb = K >> 6;               // number of 64-wide K chunks
    stage(0, 0);                          // prologue prefetch

    for (int it = 0; it < nkb; ++it) {
        const int cur = it & 1;
        const bool more = (it + 1) < nkb;
        if (more) stage(1 - cur, (it + 1) << 6);          // prefetch next chunk
        if (more) asm volatile("s_wait_asynccnt 12" ::: "memory");
        else      asm volatile("s_wait_asynccnt 0"  ::: "memory");
        __syncthreads();                                  // cur buffer visible to all waves

        const unsigned short* A = smem + cur * BUF_STRIDE;
        const unsigned short* B = A + ABUF_ELEMS;
#pragma unroll
        for (int ks = 0; ks < 2; ++ks) {                  // two K=32 sub-steps
            Frag32B b[4];
#pragma unroll
            for (int nt = 0; nt < 4; ++nt) {              // B frag: lane = col r, run at 16h
                const unsigned short* p =
                    &B[((wn * 64 + nt * 16 + r) << 6) + ks * 32 + 16 * h];
                b[nt].q[0] = *(const v4u*)p;
                b[nt].q[1] = *(const v4u*)(p + 8);
            }
#pragma unroll
            for (int mt = 0; mt < 4; ++mt) {
                Frag32B a;                                // A frag: row r, run at 16h
                const unsigned short* p =
                    &A[((wm * 64 + mt * 16 + r) << 6) + ks * 32 + 16 * h];
                a.q[0] = *(const v4u*)p;
                a.q[1] = *(const v4u*)(p + 8);
#pragma unroll
                for (int nt = 0; nt < 4; ++nt)
                    acc[mt][nt] = __builtin_amdgcn_wmma_f32_16x16x32_bf16(
                        false, a.bf, false, b[nt].bf,
                        (short)0, acc[mt][nt], false, false);
            }
        }
        __syncthreads();                  // everyone done reading cur before it is re-staged
    }

    // ---- epilogue: C layout is VGPR p -> row 8h+p, col r; add bias ----------
    const int row0 = m0 + wm * 64;
    const int col0 = n0 + wn * 64;
#pragma unroll
    for (int mt = 0; mt < 4; ++mt) {
#pragma unroll
        for (int nt = 0; nt < 4; ++nt) {
            int col  = col0 + nt * 16 + r;
            float bv = bias[col];
#pragma unroll
            for (int p = 0; p < 8; ++p) {
                int row = row0 + mt * 16 + h * 8 + p;
                out[(size_t)row * N + col] = acc[mt][nt][p] + bv;
            }
        }
    }
}

// ---- Host launcher ----------------------------------------------------------
extern "C" void kernel_launch(void* const* d_in, const int* in_sizes, int n_in,
                              void* d_out, int out_size, void* d_ws, size_t ws_size,
                              hipStream_t stream) {
    const float* x      = (const float*)d_in[0];   // [B,S,K] f32
    const float* weight = (const float*)d_in[1];   // [N,K]   f32
    const float* bias   = (const float*)d_in[2];   // [N]     f32
    const float* scores = (const float*)d_in[3];   // [N/32,K/32] f32

    const int N = in_sizes[2];               // 4096
    const int K = in_sizes[1] / N;           // 4096
    const int M = in_sizes[0] / K;           // 8192

    unsigned short* wbuf = (unsigned short*)d_ws;      // N*K bf16 (32 MB)
    unsigned short* xbuf = wbuf + (size_t)N * K;       // M*K bf16 (64 MB)

    {   // 1) fused mask + convert W
        int threads = (N * K) / 8;
        prune_convert_w<<<dim3(threads / 256), dim3(256), 0, stream>>>(
            weight, scores, wbuf, K, K / 32);
    }
    {   // 2) convert x
        int threads = (M * K) / 8;
        convert_x<<<dim3(threads / 256), dim3(256), 0, stream>>>(x, xbuf);
    }
    // 3) WMMA GEMM + bias (tile 256x128, double-buffered async staging)
    wmma_gemm<<<dim3(N / 128, M / 256), dim3(256), 0, stream>>>(
        xbuf, wbuf, bias, (float*)d_out, M, N, K);
}
